// Model_18631568130382
// MI455X (gfx1250) — matrix operand order repaired and checked by reading, not compiled, not run on hardware
//
#include <hip/hip_runtime.h>

typedef __attribute__((ext_vector_type(16))) _Float16 v16h;
typedef __attribute__((ext_vector_type(8)))  _Float16 v8h;
typedef __attribute__((ext_vector_type(8)))  float    v8f;

// ---------------------------------------------------------------------------
// Generic implicit-GEMM conv via WMMA f32 = f16 x f16 + f32
// out[m, n] = sum_k A[m,k] * B[k,n],  m = (b, oy, ox), n = oc, k = ic*T + tap
// MODE 0: normal conv (STRIDE/PAD/KW), MODE 1: transpose-conv parity class
// Block: 256 threads (8 waves), tile M=64 x N=64, K-step 32.
// Wave (mw = w&3, nw = w>>2) owns a 16x32 C tile -> 2 WMMAs sharing A-frag.
// ---------------------------------------------------------------------------
struct ConvP {
    const _Float16* __restrict__ in;    // NCHW f16
    const _Float16* __restrict__ Bw;    // packed [Kpad, OCpad] f16
    const float*    __restrict__ bias;  // [OC]
    _Float16*       __restrict__ out_h; // NCHW f16 (or null)
    float*          __restrict__ out_f; // NCHW f32 (or null)
    int Bn, IC, IH, IW;
    int OC, OCpad, OH, OW;              // OH/OW = per-parity dims in MODE 1
    int K, Kpad;
    int po, pw;                         // parity class for MODE 1
    int OHf, OWf;                       // full output dims for the store
    int relu;
};

template <int MODE, int T, int KW, int STRIDE, int PAD>
__global__ __launch_bounds__(256) void conv_gemm_wmma(ConvP p)
{
    __shared__ __align__(64) _Float16 As[64][32];   // M x K tile (4 KB)
    __shared__ __align__(64) _Float16 BsT[64][32];  // N x K tile, transposed (4 KB)

    const int tid  = threadIdx.x;
    const int lane = tid & 31;
    const int wave = tid >> 5;
    const int m0 = blockIdx.x * 64;
    const int n0 = blockIdx.y * 64;
    const int OHW = p.OH * p.OW;
    const int M = p.Bn * OHW;
    const int IHW = p.IH * p.IW;

    // A-tile staging: row ar, 8 contiguous K entries at ac0
    const int ar  = tid >> 2;
    const int ac0 = (tid & 3) * 8;
    const int am  = m0 + ar;
    const int amc = (am < M) ? am : (M - 1);
    const bool am_ok = (am < M);
    const int ab = amc / OHW;
    const int arem = amc - ab * OHW;
    const int aoy = arem / p.OW;
    const int aox = arem - aoy * p.OW;
    // Per-thread hoisted bases
    const _Float16* __restrict__ inb = p.in + (size_t)ab * p.IC * IHW;
    const int ay0 = (MODE == 0) ? (aoy * STRIDE - PAD) : aoy;
    const int ax0 = (MODE == 0) ? (aox * STRIDE - PAD) : aox;
    // MODE 1 tap offsets (fixed per parity class)
    const int diy0 = (p.po == 0) ? 0 : 1, diy1 = (p.po == 0) ? -1 : 0;
    const int dix0 = (p.pw == 0) ? 0 : 1, dix1 = (p.pw == 0) ? -1 : 0;

    // B-tile staging: column bn, 8 contiguous K entries at bkq
    const int bn  = tid & 63;
    const int bkq = (tid >> 6) * 8;

    union { v8f v; float f[8]; } acc0, acc1;
    #pragma unroll
    for (int i = 0; i < 8; i++) { acc0.f[i] = 0.f; acc1.f[i] = 0.f; }

    const int mw = wave & 3;
    const int nw = wave >> 2;
    const int hi = (lane >> 4) & 1;
    const int l15 = lane & 15;

    for (int k0 = 0; k0 < p.Kpad; k0 += 32) {
        // ---- stage A tile (im2col on the fly): branchless clamped gathers ----
        union { v8h v; _Float16 h[8]; } ast;
        #pragma unroll
        for (int j = 0; j < 8; j++) {
            int k  = k0 + ac0 + j;
            int kk = (k < p.K) ? k : (p.K - 1);   // clamp for safe addressing
            int ic = kk / T;                      // T is a compile-time constant
            int t  = kk - ic * T;
            int iy, ix;
            if (MODE == 0) {
                int kh = t / KW, kw = t - kh * KW;
                iy = ay0 + kh;
                ix = ax0 + kw;
            } else {
                int th = t >> 1, tw = t & 1;
                iy = ay0 + (th ? diy1 : diy0);
                ix = ax0 + (tw ? dix1 : dix0);
            }
            bool ok = am_ok && (k < p.K) &&
                      (iy >= 0) && (iy < p.IH) && (ix >= 0) && (ix < p.IW);
            int iyc = min(max(iy, 0), p.IH - 1);
            int ixc = min(max(ix, 0), p.IW - 1);
            _Float16 v = inb[(size_t)ic * IHW + iyc * p.IW + ixc];  // always valid
            ast.h[j] = ok ? v : (_Float16)0.f;
        }
        *(v8h*)&As[ar][ac0] = ast.v;

        // ---- stage B tile transposed: BsT[n][k] ----
        union { v8h v; _Float16 h[8]; } bst;
        #pragma unroll
        for (int j = 0; j < 8; j++)
            bst.h[j] = p.Bw[(size_t)(k0 + bkq + j) * p.OCpad + n0 + bn];
        *(v8h*)&BsT[bn][bkq] = bst.v;
        if (k0 + 32 < p.Kpad)
            __builtin_prefetch(p.Bw + (size_t)(k0 + 32 + bkq) * p.OCpad + n0 + bn, 0, 1);
        __syncthreads();

        // ---- fragments (k = 16*hi + s, same in A and B): contiguous LDS b128 reads ----
        const int arow = mw * 16 + l15;
        v8h a_lo = *(const v8h*)&As[arow][16 * hi];
        v8h a_hi = *(const v8h*)&As[arow][16 * hi + 8];
        v16h af = __builtin_shufflevector(a_lo, a_hi,
                    0,1,2,3,4,5,6,7,8,9,10,11,12,13,14,15);
        v16h bf0 = *(const v16h*)&BsT[nw * 32 + l15][16 * hi];
        v16h bf1 = *(const v16h*)&BsT[nw * 32 + 16 + l15][16 * hi];

        acc0.v = __builtin_amdgcn_wmma_f32_16x16x32_f16(
                     false, af, false, bf0, (short)0, acc0.v, false, false);
        acc1.v = __builtin_amdgcn_wmma_f32_16x16x32_f16(
                     false, af, false, bf1, (short)0, acc1.v, false, false);
        __syncthreads();
    }

    // ---- epilogue: bias (+relu), scatter to NCHW (parity-interleaved MODE 1)
    #pragma unroll
    for (int half = 0; half < 2; half++) {
        const float* accf = half ? acc1.f : acc0.f;
        const int n = n0 + nw * 32 + half * 16 + l15;
        #pragma unroll
        for (int r = 0; r < 8; r++) {
            int mm = m0 + mw * 16 + r + 8 * hi;
            if (mm < M && n < p.OC) {
                float v = accf[r] + p.bias[n];
                if (p.relu) v = fmaxf(v, 0.f);
                int b_ = mm / OHW;
                int rem = mm - b_ * OHW;
                int oy = rem / p.OW, ox = rem - oy * p.OW;
                int oyf = oy, oxf = ox;
                if (MODE == 1) { oyf = 2 * oy + p.po; oxf = 2 * ox + p.pw; }
                size_t idx = (((size_t)b_ * p.OC + n) * p.OHf + oyf) * p.OWf + oxf;
                if (p.out_h) p.out_h[idx] = (_Float16)v;
                if (p.out_f) p.out_f[idx] = v;
            }
        }
    }
}

// ---------------------------------------------------------------------------
// Weight packers: produce f16 [Kpad, OCpad] GEMM-ready B matrices
// ---------------------------------------------------------------------------
__global__ void pack_conv_w(const float* __restrict__ w, _Float16* __restrict__ Bw,
                            int OC, int IC, int KHW, int Kpad, int OCpad)
{
    int i = blockIdx.x * 256 + threadIdx.x;
    if (i >= Kpad * OCpad) return;
    int k = i / OCpad, n = i - k * OCpad;
    float v = 0.f;
    int K = IC * KHW;
    if (k < K && n < OC) {
        int ic = k / KHW, t = k - ic * KHW;           // t = kh*KW + kw (contiguous)
        v = w[((size_t)n * IC + ic) * KHW + t];       // OIHW
    }
    Bw[i] = (_Float16)v;
}

// ConvTranspose2d(k=4,s=2,p=1): parity class (po,pw) uses taps
//   po==0: (kh=1,dih=0),(kh=3,dih=-1);  po==1: (kh=0,dih=+1),(kh=2,dih=0)
__global__ void pack_convt_w(const float* __restrict__ w, _Float16* __restrict__ Bw,
                             int IC, int OC, int po, int pw, int OCpad)
{
    int Kpad = IC * 4;
    int i = blockIdx.x * 256 + threadIdx.x;
    if (i >= Kpad * OCpad) return;
    int k = i / OCpad, n = i - k * OCpad;
    float v = 0.f;
    if (n < OC) {
        int ic = k >> 2, t = k & 3;
        int th = t >> 1, tw = t & 1;
        int kh = po ? (th ? 2 : 0) : (th ? 3 : 1);
        int kw = pw ? (tw ? 2 : 0) : (tw ? 3 : 1);
        v = w[(((size_t)ic * OC + n) * 4 + kh) * 4 + kw];   // [in, out, 4, 4]
    }
    Bw[i] = (_Float16)v;
}

// ---------------------------------------------------------------------------
// Nearest resize 224 -> 256 (idx = floor(i*224/256) = (7*i)>>3); f32 + f16 copies
// ---------------------------------------------------------------------------
__global__ void resize_nearest(const float* __restrict__ imgs,
                               float* __restrict__ xf, _Float16* __restrict__ xh)
{
    int i = blockIdx.x * 256 + threadIdx.x;
    if (i >= 16 * 3 * 256 * 256) return;
    int ox = i & 255;
    int oy = (i >> 8) & 255;
    int t  = i >> 16;                       // b*3 + c
    int iy = (oy * 7) >> 3;
    int ix = (ox * 7) >> 3;
    float v = imgs[((size_t)t * 224 + iy) * 224 + ix];
    xf[i] = v;
    xh[i] = (_Float16)v;
}

// ---------------------------------------------------------------------------
// Vector quantization: argmin over 512 codes (f32, LDS-tiled), writes ids,
// quantized f16 feature map (straight-through => q == quant), and diff partials
// ---------------------------------------------------------------------------
__global__ __launch_bounds__(256) void vq_kernel(const float* __restrict__ z,
                                                 const float* __restrict__ embed,
                                                 int* __restrict__ ids,
                                                 _Float16* __restrict__ q,
                                                 float* __restrict__ partials)
{
    __shared__ float Es[64][64];
    __shared__ float esq[64];
    __shared__ float red[256];
    int tid = threadIdx.x;
    int pix = blockIdx.x * 256 + tid;       // 0..65535  (b*4096 + h*64 + w)
    int b  = pix >> 12;
    int hw = pix & 4095;

    float ze[64];
    #pragma unroll
    for (int c = 0; c < 64; c++)
        ze[c] = z[(((size_t)b * 64 + c) << 12) + hw];

    float best = 3.4e38f; int bid = 0;
    for (int e0 = 0; e0 < 512; e0 += 64) {
        __syncthreads();
        #pragma unroll
        for (int i = 0; i < 16; i++) {      // 4096 elems / 256 threads
            int idx = tid + i * 256;
            int c = idx >> 6, e = idx & 63;
            Es[c][e] = embed[(size_t)c * 512 + e0 + e];
        }
        __syncthreads();
        if (tid < 64) {
            float s = 0.f;
            #pragma unroll
            for (int c = 0; c < 64; c++) { float v = Es[c][tid]; s += v * v; }
            esq[tid] = s;
        }
        __syncthreads();
        for (int e = 0; e < 64; e++) {
            float dot = 0.f;
            #pragma unroll
            for (int c = 0; c < 64; c++) dot += ze[c] * Es[c][e];
            float d = esq[e] - 2.f * dot;   // ze^2 term is constant per pixel
            if (d < best) { best = d; bid = e0 + e; }
        }
    }
    ids[pix] = bid;

    float diffp = 0.f;
    #pragma unroll
    for (int c = 0; c < 64; c++) {
        float ev = embed[(size_t)c * 512 + bid];
        float d  = ev - ze[c];
        diffp += d * d;
        q[(((size_t)b * 64 + c) << 12) + hw] = (_Float16)ev;
    }
    red[tid] = diffp;
    __syncthreads();
    for (int s = 128; s > 0; s >>= 1) {
        if (tid < s) red[tid] += red[tid + s];
        __syncthreads();
    }
    if (tid == 0) partials[blockIdx.x] = red[0];
}

__global__ void reduce_diff(const float* __restrict__ partials, float* __restrict__ out)
{
    __shared__ float red[256];
    int tid = threadIdx.x;
    red[tid] = partials[tid];
    __syncthreads();
    for (int s = 128; s > 0; s >>= 1) {
        if (tid < s) red[tid] += red[tid + s];
        __syncthreads();
    }
    if (tid == 0) out[0] = red[0] / (16.f * 64.f * 64.f * 64.f);
}

// ---------------------------------------------------------------------------
extern "C" void kernel_launch(void* const* d_in, const int* in_sizes, int n_in,
                              void* d_out, int out_size, void* d_ws, size_t ws_size,
                              hipStream_t stream)
{
    (void)in_sizes; (void)n_in; (void)out_size; (void)ws_size;
    const float* imgs  = (const float*)d_in[0];
    const float* ew1   = (const float*)d_in[1];
    const float* eb1   = (const float*)d_in[2];
    const float* ew2   = (const float*)d_in[3];
    const float* eb2   = (const float*)d_in[4];
    const float* ew3   = (const float*)d_in[5];
    const float* eb3   = (const float*)d_in[6];
    const float* emb   = (const float*)d_in[7];
    const float* dw1   = (const float*)d_in[8];
    const float* db1   = (const float*)d_in[9];
    const float* dw2   = (const float*)d_in[10];
    const float* db2   = (const float*)d_in[11];
    const float* dw3   = (const float*)d_in[12];
    const float* db3   = (const float*)d_in[13];

    // d_out: product[3145728] | x[3145728] | ids[65536 int] | diff[1]
    float* out  = (float*)d_out;
    float* prod = out;
    float* xout = out + 3145728;
    int*   ids  = (int*)(out + 6291456);
    float* diff = out + 6356992;

    // workspace carve-out (aliased: bigA = h1/d2, bufB = h2/d1)
    char* ws = (char*)d_ws;
    size_t off = 0;
    auto carve = [&](size_t bytes) -> char* {
        char* p = ws + off; off += (bytes + 255) & ~(size_t)255; return p;
    };
    _Float16* xh   = (_Float16*)carve(6291456);         // x f16
    _Float16* bigA = (_Float16*)carve(67108864);        // h1 [16,128,128,128] then d2
    _Float16* bufB = (_Float16*)carve(16777216);        // h2 [16,128,64,64] then d1
    float*    z    = (float*)   carve(16777216);        // z  [16,64,64,64] f32
    _Float16* q    = (_Float16*)carve(8388608);         // quant f16
    _Float16* w1p  = (_Float16*)carve(64   * 128 * 2);
    _Float16* w2p  = (_Float16*)carve(2048 * 128 * 2);
    _Float16* w3p  = (_Float16*)carve(1152 * 64  * 2);
    _Float16* wd1p = (_Float16*)carve(576  * 128 * 2);
    _Float16* wt2p = (_Float16*)carve(4 * 512 * 128 * 2);
    _Float16* wt3p = (_Float16*)carve(4 * 512 * 64  * 2);
    float* partials = (float*)carve(256 * 4);

    // ---- pack weights ----
    pack_conv_w<<<(64*128+255)/256,   256, 0, stream>>>(ew1, w1p, 128, 3,   16, 64,   128);
    pack_conv_w<<<(2048*128+255)/256, 256, 0, stream>>>(ew2, w2p, 128, 128, 16, 2048, 128);
    pack_conv_w<<<(1152*64+255)/256,  256, 0, stream>>>(ew3, w3p, 64,  128, 9,  1152, 64);
    pack_conv_w<<<(576*128+255)/256,  256, 0, stream>>>(dw1, wd1p, 128, 64, 9,  576,  128);
    for (int pc = 0; pc < 4; pc++) {
        pack_convt_w<<<(512*128+255)/256, 256, 0, stream>>>(dw2, wt2p + pc*512*128, 128, 128, pc>>1, pc&1, 128);
        pack_convt_w<<<(512*64 +255)/256, 256, 0, stream>>>(dw3, wt3p + pc*512*64,  128, 3,   pc>>1, pc&1, 64);
    }

    // ---- resize ----
    resize_nearest<<<12288, 256, 0, stream>>>(imgs, xout, xh);

    // ---- encoder ----
    ConvP c1 = { xh, w1p, eb1, bigA, nullptr, 16, 3, 256, 256, 128, 128, 128, 128,
                 48, 64, 0, 0, 128, 128, 1 };
    conv_gemm_wmma<0,16,4,2,1><<<dim3(4096, 2), 256, 0, stream>>>(c1);

    ConvP c2 = { bigA, w2p, eb2, bufB, nullptr, 16, 128, 128, 128, 128, 128, 64, 64,
                 2048, 2048, 0, 0, 64, 64, 1 };
    conv_gemm_wmma<0,16,4,2,1><<<dim3(1024, 2), 256, 0, stream>>>(c2);

    ConvP c3 = { bufB, w3p, eb3, nullptr, z, 16, 128, 64, 64, 64, 64, 64, 64,
                 1152, 1152, 0, 0, 64, 64, 0 };
    conv_gemm_wmma<0,9,3,1,1><<<dim3(1024, 1), 256, 0, stream>>>(c3);

    // ---- vector quantization ----
    vq_kernel<<<256, 256, 0, stream>>>(z, emb, ids, q, partials);
    reduce_diff<<<1, 256, 0, stream>>>(partials, diff);

    // ---- decoder ----
    ConvP dc1 = { q, wd1p, db1, bufB, nullptr, 16, 64, 64, 64, 128, 128, 64, 64,
                  576, 576, 0, 0, 64, 64, 1 };
    conv_gemm_wmma<0,9,3,1,1><<<dim3(1024, 2), 256, 0, stream>>>(dc1);

    for (int pc = 0; pc < 4; pc++) {
        ConvP dt2 = { bufB, wt2p + pc*512*128, db2, bigA, nullptr, 16, 128, 64, 64,
                      128, 128, 64, 64, 512, 512, pc>>1, pc&1, 128, 128, 1 };
        conv_gemm_wmma<1,4,1,1,0><<<dim3(1024, 2), 256, 0, stream>>>(dt2);
    }
    for (int pc = 0; pc < 4; pc++) {
        ConvP dt3 = { bigA, wt3p + pc*512*64, db3, nullptr, prod, 16, 128, 128, 128,
                      3, 64, 128, 128, 512, 512, pc>>1, pc&1, 256, 256, 0 };
        conv_gemm_wmma<1,4,1,1,0><<<dim3(4096, 1), 256, 0, stream>>>(dt3);
    }
}
